// MultiHeadSelfAttention_86827058856509
// MI455X (gfx1250) — matrix operand order repaired and checked by reading, compiled, not run on hardware
//
#include <hip/hip_runtime.h>

#define NTOK 49
#define NPAD 64
#define CDIM 512
#define HEADS 8
#define HD 64
#define LDT 520   // stride (f16) for [token][C] buffers: 1040B -> 4-bank lane skew
#define LDQ 72    // stride (f16) for 64x64 q/k/v tiles
#define LDA 68    // stride (f32) for attention scores/probs

typedef __attribute__((ext_vector_type(16))) _Float16 v16h;
typedef __attribute__((ext_vector_type(8)))  _Float16 v8h;
typedef __attribute__((ext_vector_type(8)))  float    v8f;

#define WMMA(acc, a, b)                                                        \
  acc = __builtin_amdgcn_wmma_f32_16x16x32_f16(false, (a), false, (b),         \
                                               (short)0, (acc), false, false)

__device__ __forceinline__ v16h cat8(v8h lo, v8h hi) {
  return __builtin_shufflevector(lo, hi, 0, 1, 2, 3, 4, 5, 6, 7, 8, 9, 10, 11,
                                 12, 13, 14, 15);
}

// 16 consecutive f16 at p (B-fragment / K-contiguous operand, kb = kk*32 +
// khalf*16).  Two ds_load_b128.
__device__ __forceinline__ v16h frag_f16_k16(const _Float16* p) {
  const v8h* q = reinterpret_cast<const v8h*>(p);
  return cat8(q[0], q[1]);
}

// A-fragment from f16 LDS, K contiguous per row: [kb..kb+15]
// (kb = kk*32 + khalf*16).  Two ds_load_b128.
__device__ __forceinline__ v16h frag_f16_a8(const _Float16* p) {
  const v8h* q = reinterpret_cast<const v8h*>(p);
  return cat8(q[0], q[1]);
}

// A-fragment from row-major f32 global weights; kb = kk*32 + khalf*16.
__device__ __forceinline__ v16h frag_a_global_f32(const float* Wrow, int kb) {
  const float4* p = reinterpret_cast<const float4*>(Wrow + kb);
  float4 f0 = p[0], f1 = p[1], f2 = p[2], f3 = p[3];
  v16h a;
  a[0]  = (_Float16)f0.x; a[1]  = (_Float16)f0.y; a[2]  = (_Float16)f0.z; a[3]  = (_Float16)f0.w;
  a[4]  = (_Float16)f1.x; a[5]  = (_Float16)f1.y; a[6]  = (_Float16)f1.z; a[7]  = (_Float16)f1.w;
  a[8]  = (_Float16)f2.x; a[9]  = (_Float16)f2.y; a[10] = (_Float16)f2.z; a[11] = (_Float16)f2.w;
  a[12] = (_Float16)f3.x; a[13] = (_Float16)f3.y; a[14] = (_Float16)f3.z; a[15] = (_Float16)f3.w;
  return a;
}

// B-fragment from 16 consecutive f32 in LDS (kb = kk*32 + khalf*16).
__device__ __forceinline__ v16h frag_b_f32_k16(const float* p) {
  const float4* q = reinterpret_cast<const float4*>(p);
  float4 f0 = q[0], f1 = q[1], f2 = q[2], f3 = q[3];
  v16h b;
  b[0]  = (_Float16)f0.x; b[1]  = (_Float16)f0.y; b[2]  = (_Float16)f0.z; b[3]  = (_Float16)f0.w;
  b[4]  = (_Float16)f1.x; b[5]  = (_Float16)f1.y; b[6]  = (_Float16)f1.z; b[7]  = (_Float16)f1.w;
  b[8]  = (_Float16)f2.x; b[9]  = (_Float16)f2.y; b[10] = (_Float16)f2.z; b[11] = (_Float16)f2.w;
  b[12] = (_Float16)f3.x; b[13] = (_Float16)f3.y; b[14] = (_Float16)f3.z; b[15] = (_Float16)f3.w;
  return b;
}

// Convert 8 accumulator f32 lanes to f16 and store as one ds_store_b128.
__device__ __forceinline__ void store8(_Float16* p, v8f a) {
  v8h r;
#pragma unroll
  for (int i = 0; i < 8; ++i) r[i] = (_Float16)a[i];
  *reinterpret_cast<v8h*>(p) = r;
}

__global__ __launch_bounds__(256) void mhsa_win_attn_kernel(
    const float* __restrict__ x, const float* __restrict__ qkv_w,
    const float* __restrict__ proj_w, const float* __restrict__ proj_b,
    const float* __restrict__ bias_table, const int* __restrict__ rel_index,
    float* __restrict__ out) {
  __shared__ _Float16 XhT[NPAD * LDT];    // x[b]^T   : [token][C]
  __shared__ _Float16 OutAT[NPAD * LDT];  // attnout^T: [token][C]
  __shared__ _Float16 QsT[NPAD * LDQ];    // q^T : [token][d]
  __shared__ _Float16 KsT[NPAD * LDQ];    // k^T : [token][d]
  __shared__ _Float16 Vs[NPAD * LDQ];     // v   : [d][token]
  __shared__ float    At[NPAD * LDA];     // scores/probs [query][key]

  const int tid   = threadIdx.x;
  const int lane  = tid & 31;
  const int wave  = tid >> 5;
  const int lm    = lane & 15;
  const int khalf = lane >> 4;
  const int bidx  = blockIdx.x;

  // ---- Stage 1: x[b] (f32 [C][49]) -> XhT (f16 [tok][C]); zero pad rows ----
  const float* xb = x + (size_t)bidx * CDIM * NTOK;
  for (int i = tid; i < CDIM * NTOK; i += 256) {
    int c = i / NTOK, n = i - c * NTOK;
    XhT[n * LDT + c] = (_Float16)xb[i];
  }
  for (int i = tid; i < (NPAD - NTOK) * CDIM; i += 256) {
    int r = NTOK + i / CDIM, c = i - (r - NTOK) * CDIM;
    XhT[r * LDT + c] = (_Float16)0.f;
  }
  __syncthreads();

  for (int hh = 0; hh < HEADS; ++hh) {
    // ---- 2a: q/k/v = Wqkv[head] (64x512) @ X; 24 strips, 2 accs each ----
    for (int s = wave; s < 24; s += 8) {
      int mat = s >> 3, r = s & 7, mt = r >> 1, nh = r & 1;
      const float* Wrow =
          qkv_w + (size_t)(mat * CDIM + hh * HD + mt * 16 + lm) * CDIM;
      int c0 = nh * 32 + lm, c1 = c0 + 16;
      v8f a0 = {}, a1 = {};
#pragma unroll 4
      for (int kk = 0; kk < 16; ++kk) {
        v16h af = frag_a_global_f32(Wrow, kk * 32 + khalf * 16);
        int kb = kk * 32 + khalf * 16;
        v16h b0 = frag_f16_k16(XhT + c0 * LDT + kb);
        v16h b1 = frag_f16_k16(XhT + c1 * LDT + kb);
        WMMA(a0, af, b0);
        WMMA(a1, af, b1);
      }
      if (mat == 2) {  // V stored [d][token] (strided stores; wide loads later)
#pragma unroll
        for (int j = 0; j < 8; ++j) {
          int dr = mt * 16 + j + khalf * 8;
          Vs[dr * LDQ + c0] = (_Float16)a0[j];
          Vs[dr * LDQ + c1] = (_Float16)a1[j];
        }
      } else {  // Q,K stored transposed [token][d]: contiguous b128 stores
        _Float16* T = (mat == 0) ? QsT : KsT;
        int dbase = mt * 16 + khalf * 8;
        store8(T + c0 * LDQ + dbase, a0);
        store8(T + c1 * LDQ + dbase, a1);
      }
    }
    __syncthreads();

    // ---- 2b: S^T = K @ Q^T (=> At[query][key]), *scale + bias ----
    {
      int kt = wave & 3, qh = wave >> 2;
      int keyrow = kt * 16 + lm;            // A row (key token)
      int q0 = qh * 32 + lm, q1 = q0 + 16;  // B cols (query tokens)
      v8f a0 = {}, a1 = {};
#pragma unroll
      for (int kk = 0; kk < 2; ++kk) {
        v16h af = frag_f16_a8(KsT + keyrow * LDQ + kk * 32 + khalf * 16);
        int kb = kk * 32 + khalf * 16;
        v16h b0 = frag_f16_k16(QsT + q0 * LDQ + kb);
        v16h b1 = frag_f16_k16(QsT + q1 * LDQ + kb);
        WMMA(a0, af, b0);
        WMMA(a1, af, b1);
      }
      int kbase = kt * 16 + khalf * 8;
      float s0[8], s1[8];
#pragma unroll
      for (int j = 0; j < 8; ++j) {
        int key = kbase + j;
        float v0 = a0[j] * 0.125f;  // d^-0.5, d = 64
        float v1 = a1[j] * 0.125f;
        if (key < NTOK) {
          if (q0 < NTOK) v0 += bias_table[rel_index[q0 * NTOK + key] * HEADS + hh];
          if (q1 < NTOK) v1 += bias_table[rel_index[q1 * NTOK + key] * HEADS + hh];
        }
        s0[j] = v0;
        s1[j] = v1;
      }
      float4* p0 = reinterpret_cast<float4*>(At + q0 * LDA + kbase);
      p0[0] = make_float4(s0[0], s0[1], s0[2], s0[3]);
      p0[1] = make_float4(s0[4], s0[5], s0[6], s0[7]);
      float4* p1 = reinterpret_cast<float4*>(At + q1 * LDA + kbase);
      p1[0] = make_float4(s1[0], s1[1], s1[2], s1[3]);
      p1[1] = make_float4(s1[4], s1[5], s1[6], s1[7]);
    }
    __syncthreads();

    // ---- 2c: row softmax; zero K-padding so 0*garbage never NaNs ----
    if (tid < NPAD) {
      int r = tid;
      if (r < NTOK) {
        float mx = -3.4e38f;
        for (int m = 0; m < NTOK; ++m) mx = fmaxf(mx, At[r * LDA + m]);
        float sum = 0.f;
        for (int m = 0; m < NTOK; ++m) {
          float e = __expf(At[r * LDA + m] - mx);
          At[r * LDA + m] = e;
          sum += e;
        }
        float inv = 1.f / sum;
        for (int m = 0; m < NTOK; ++m) At[r * LDA + m] *= inv;
        for (int m = NTOK; m < NPAD; ++m) At[r * LDA + m] = 0.f;
      } else {
        for (int m = 0; m < NPAD; ++m) At[r * LDA + m] = 0.f;
      }
    }
    __syncthreads();

    // ---- 2d: out_h^T = V @ P^T -> OutAT[token][hh*64+dd], b128 stores ----
    {
      int dt = wave & 3, th = wave >> 2;
      int drow = dt * 16 + lm;              // A row (dd)
      int t0 = th * 32 + lm, t1 = t0 + 16;  // B cols (tokens)
      v8f a0 = {}, a1 = {};
#pragma unroll
      for (int kk = 0; kk < 2; ++kk) {
        v16h af = frag_f16_a8(Vs + drow * LDQ + kk * 32 + khalf * 16);
        int kb = kk * 32 + khalf * 16;
        v16h b0 = frag_b_f32_k16(At + t0 * LDA + kb);
        v16h b1 = frag_b_f32_k16(At + t1 * LDA + kb);
        WMMA(a0, af, b0);
        WMMA(a1, af, b1);
      }
      int dbase = hh * HD + dt * 16 + khalf * 8;
      store8(OutAT + t0 * LDT + dbase, a0);
      store8(OutAT + t1 * LDT + dbase, a1);
    }
    __syncthreads();
  }

  // ---- Stage 3: out = proj_w @ attnout + proj_b; 32 strips, 4 accs ----
  float* ob = out + (size_t)bidx * CDIM * NTOK;
  for (int s = wave; s < 32; s += 8) {
    const float* Wrow = proj_w + (size_t)(s * 16 + lm) * CDIM;
    int c0 = lm, c1 = 16 + lm, c2 = 32 + lm, c3 = 48 + lm;
    v8f a0 = {}, a1 = {}, a2 = {}, a3 = {};
#pragma unroll 4
    for (int kk = 0; kk < 16; ++kk) {
      v16h af = frag_a_global_f32(Wrow, kk * 32 + khalf * 16);
      int kb = kk * 32 + khalf * 16;
      v16h b0 = frag_f16_k16(OutAT + c0 * LDT + kb);
      v16h b1 = frag_f16_k16(OutAT + c1 * LDT + kb);
      v16h b2 = frag_f16_k16(OutAT + c2 * LDT + kb);
      v16h b3 = frag_f16_k16(OutAT + c3 * LDT + kb);
      WMMA(a0, af, b0);
      WMMA(a1, af, b1);
      WMMA(a2, af, b2);
      WMMA(a3, af, b3);
    }
#pragma unroll
    for (int j = 0; j < 8; ++j) {
      int gr = s * 16 + j + khalf * 8;
      float pb = proj_b[gr];
      ob[gr * NTOK + c0] = a0[j] + pb;  // c0 <= 15
      ob[gr * NTOK + c1] = a1[j] + pb;  // c1 <= 31
      ob[gr * NTOK + c2] = a2[j] + pb;  // c2 <= 47
      if (c3 < NTOK) ob[gr * NTOK + c3] = a3[j] + pb;
    }
  }
}

extern "C" void kernel_launch(void* const* d_in, const int* in_sizes, int n_in,
                              void* d_out, int out_size, void* d_ws,
                              size_t ws_size, hipStream_t stream) {
  (void)in_sizes; (void)n_in; (void)out_size; (void)d_ws; (void)ws_size;
  const float* x          = (const float*)d_in[0];
  const float* qkv_w      = (const float*)d_in[1];
  const float* proj_w     = (const float*)d_in[2];
  const float* proj_b     = (const float*)d_in[3];
  const float* bias_table = (const float*)d_in[4];
  const int*   rel_index  = (const int*)d_in[5];
  float* out = (float*)d_out;

  dim3 grid(2048);
  dim3 block(256);
  mhsa_win_attn_kernel<<<grid, block, 0, stream>>>(x, qkv_w, proj_w, proj_b,
                                                   bias_table, rel_index, out);
}
